// MultiHeadAttention_6725918786005
// MI455X (gfx1250) — compile-verified, matrix operand order checked by reading, not run on hardware
//
#include <hip/hip_runtime.h>
#include <hip/hip_bf16.h>

// ---------------------------------------------------------------------------
// MHA forward on gfx1250: bf16 WMMA everywhere, flash-style attention.
//   x:      [4, 2048, 1024] f32        W_qkv: [3072, 1024] f32   b_qkv: [3072]
//   W_out:  [1024, 1024] f32           b_out: [1024]
//   out:    [4, 2048, 1024] f32
// Software-pipelined fragment loads (ping-pong) so WMMAs overlap VMEM.
// ---------------------------------------------------------------------------

typedef __attribute__((ext_vector_type(16))) __bf16 v16bf;
typedef __attribute__((ext_vector_type(8)))  float  v8f;

union BFrag { v16bf v; unsigned int u[8]; };

#define LOG2E 1.44269504088896340736f

__device__ __forceinline__ unsigned short f2bf(float f) {
  union { float f; unsigned int u; } c; c.f = f;
  unsigned int u = c.u;
  unsigned int r = u + 0x7FFFu + ((u >> 16) & 1u);   // round-to-nearest-even
  return (unsigned short)(r >> 16);
}

__device__ __forceinline__ unsigned int pack2bf(float lo, float hi) {
  return (unsigned int)f2bf(lo) | ((unsigned int)f2bf(hi) << 16);
}

// Load one 16x32 bf16 A/B fragment from a row-major matrix.
// `p` points at (this lane's row) + k-chunk base; h = lane/16.
// ISA 16-bit layout: vgpr v holds K pair k0 = 16*(v/4) + 2*(v&3) + 8*h.
__device__ __forceinline__ v16bf load_frag(const unsigned short* p, int h) {
  BFrag f;
#pragma unroll
  for (int v = 0; v < 8; ++v) {
    int k0 = ((v >> 2) << 4) + ((v & 3) << 1) + (h << 3);
    f.u[v] = *(const unsigned int*)(p + k0);
  }
  return f.v;
}

__device__ __forceinline__ v8f wmma_bf16(v16bf a, v16bf b, v8f c) {
  return __builtin_amdgcn_wmma_f32_16x16x32_bf16(false, a, false, b,
                                                 (short)0, c, false, false);
}

// ---------------------------------------------------------------------------
// fp32 -> bf16 conversion (grid-stride)
// ---------------------------------------------------------------------------
__global__ void cvt_bf16(const float* __restrict__ in,
                         unsigned short* __restrict__ out, int n) {
  int i = blockIdx.x * blockDim.x + threadIdx.x;
  int stride = gridDim.x * blockDim.x;
  for (; i < n; i += stride) out[i] = f2bf(in[i]);
}

// ---------------------------------------------------------------------------
// QKV projection: qkv[m][n] = sum_k X[m][k] * Wqkv[n][k] + b[n]
// M=8192 (B*S), K=1024, N=3072.  One wave -> 16(M) x 64(N) tile.
// K-loop unrolled x2 with ping-pong fragment buffers (loads lead WMMAs by
// one stage so v_wmma overlaps VMEM latency).
// Output scattered: Q,K token-major [bh][s][64]; V transposed [bh][64][s].
// (reference reshape is (B,S,H,3*dh): head = n/192, j = n%192; q<64,k<128,v)
// ---------------------------------------------------------------------------
__global__ void __launch_bounds__(128) gemm_qkv(
    const unsigned short* __restrict__ X, const unsigned short* __restrict__ W,
    const float* __restrict__ bias, unsigned short* __restrict__ Qb,
    unsigned short* __restrict__ Kb, unsigned short* __restrict__ Vt) {
  const int lane = threadIdx.x & 31;
  const int wave = threadIdx.x >> 5;
  const int h = lane >> 4, ln = lane & 15;
  const int mt = blockIdx.x;                       // 0..511
  const int nbase = blockIdx.y * 256 + wave * 64;  // 0..3008

  const unsigned short* arow = X + (size_t)(mt * 16 + ln) * 1024;
  const unsigned short* brow[4];
#pragma unroll
  for (int f = 0; f < 4; ++f)
    brow[f] = W + (size_t)(nbase + f * 16 + ln) * 1024;

  const v8f zero = {0.f, 0.f, 0.f, 0.f, 0.f, 0.f, 0.f, 0.f};
  v8f acc[4] = {zero, zero, zero, zero};

  // prologue: stage 0 fragments
  v16bf a0 = load_frag(arow, h);
  v16bf b0[4];
#pragma unroll
  for (int f = 0; f < 4; ++f) b0[f] = load_frag(brow[f], h);

  for (int kc = 0; kc < 1024; kc += 64) {
    __builtin_prefetch(arow + kc + 128, 0, 1);     // global_prefetch_b8
    // stage A: issue loads for kc+32, compute on kc
    v16bf a1 = load_frag(arow + kc + 32, h);
    v16bf b1[4];
#pragma unroll
    for (int f = 0; f < 4; ++f) b1[f] = load_frag(brow[f] + kc + 32, h);
#pragma unroll
    for (int f = 0; f < 4; ++f) acc[f] = wmma_bf16(a0, b0[f], acc[f]);

    // stage B: issue loads for kc+64 (wrap on last iter; redundant, harmless),
    // compute on kc+32
    const int kn = (kc + 64 < 1024) ? kc + 64 : 0;
    a0 = load_frag(arow + kn, h);
#pragma unroll
    for (int f = 0; f < 4; ++f) b0[f] = load_frag(brow[f] + kn, h);
#pragma unroll
    for (int f = 0; f < 4; ++f) acc[f] = wmma_bf16(a1, b1[f], acc[f]);
  }

#pragma unroll
  for (int f = 0; f < 4; ++f) {
    const int n = nbase + f * 16 + ln;
    const float bv = bias[n];
    const int hd = n / 192;
    const int j = n % 192;
#pragma unroll
    for (int r = 0; r < 8; ++r) {
      const int row = mt * 16 + r + 8 * h;  // global token index
      const int b_ = row >> 11;             // batch
      const int s = row & 2047;             // seq pos
      const unsigned short val = f2bf(acc[f][r] + bv);
      const int bh = b_ * 16 + hd;
      if (j < 64)
        Qb[((size_t)bh * 2048 + s) * 64 + j] = val;
      else if (j < 128)
        Kb[((size_t)bh * 2048 + s) * 64 + (j - 64)] = val;
      else
        Vt[((size_t)bh * 64 + (j - 128)) * 2048 + s] = val;
    }
  }
}

// ---------------------------------------------------------------------------
// Flash attention: one wave per (bh, 16-query tile). Streams 32 keys/iter.
//   S = Q*K^T * 0.125  -> online softmax (row stats via 16-lane shfl_xor)
//   P (C-layout) -> LDS -> A-layout bf16 -> O += P*V   (V is pre-transposed)
// V fragments are loaded right after the QK^T WMMAs so the softmax VALU work
// and the LDS transpose hide their latency.
// ---------------------------------------------------------------------------
__global__ void __launch_bounds__(128) attn(
    const unsigned short* __restrict__ Qb, const unsigned short* __restrict__ Kb,
    const unsigned short* __restrict__ Vt, unsigned short* __restrict__ Vals) {
  __shared__ float lds[4 * 16 * 32];  // 2KB per wave: P tile staging
  const int lane = threadIdx.x & 31;
  const int wave = threadIdx.x >> 5;
  const int h = lane >> 4, ln = lane & 15;
  float* pbuf = lds + wave * 512;

  const int bh = blockIdx.y;                 // 0..63  (b*16 + head)
  const int qt = blockIdx.x * 4 + wave;      // 0..127 (16-row query tile)

  const unsigned short* Q = Qb + ((size_t)bh * 2048 + qt * 16) * 64;
  const unsigned short* K = Kb + (size_t)bh * 2048 * 64;
  const unsigned short* V = Vt + (size_t)bh * 64 * 2048;

  // Q tile fragments, dh split 0..31 / 32..63
  const v16bf qf0 = load_frag(Q + ln * 64, h);
  const v16bf qf1 = load_frag(Q + ln * 64 + 32, h);

  const v8f zero = {0.f, 0.f, 0.f, 0.f, 0.f, 0.f, 0.f, 0.f};
  v8f o[4] = {zero, zero, zero, zero};  // O tile 16x64 (4 n-chunks)
  float mrow[8], lrow[8];
#pragma unroll
  for (int r = 0; r < 8; ++r) { mrow[r] = -3.0e38f; lrow[r] = 0.f; }

  for (int kc = 0; kc < 2048; kc += 32) {
    // ---- S = Q * K^T for 32 keys (two 16x16 C tiles) ----
    const unsigned short* k0p = K + (size_t)(kc + ln) * 64;
    const unsigned short* k1p = K + (size_t)(kc + 16 + ln) * 64;
    __builtin_prefetch(k0p + 32 * 64, 0, 1);
    v8f s0 = zero, s1 = zero;
    s0 = wmma_bf16(qf0, load_frag(k0p, h), s0);
    s0 = wmma_bf16(qf1, load_frag(k0p + 32, h), s0);
    s1 = wmma_bf16(qf0, load_frag(k1p, h), s1);
    s1 = wmma_bf16(qf1, load_frag(k1p + 32, h), s1);

    // ---- issue V fragment loads early (independent of softmax) ----
    v16bf vf[4];
#pragma unroll
    for (int f = 0; f < 4; ++f)
      vf[f] = load_frag(V + (size_t)(f * 16 + ln) * 2048 + kc, h);

    // ---- online softmax (per row m = r + 8*h; cols live in 16-lane group)
    float alpha[8];
#pragma unroll
    for (int r = 0; r < 8; ++r) {
      float sa = s0[r] * 0.125f, sb = s1[r] * 0.125f;
      float mx = fmaxf(sa, sb);
      mx = fmaxf(mx, __shfl_xor(mx, 1));
      mx = fmaxf(mx, __shfl_xor(mx, 2));
      mx = fmaxf(mx, __shfl_xor(mx, 4));
      mx = fmaxf(mx, __shfl_xor(mx, 8));
      const float mnew = fmaxf(mrow[r], mx);
      const float a = exp2f((mrow[r] - mnew) * LOG2E);
      const float p0 = exp2f((sa - mnew) * LOG2E);
      const float p1 = exp2f((sb - mnew) * LOG2E);
      float rs = p0 + p1;
      rs += __shfl_xor(rs, 1);
      rs += __shfl_xor(rs, 2);
      rs += __shfl_xor(rs, 4);
      rs += __shfl_xor(rs, 8);
      mrow[r] = mnew;
      alpha[r] = a;
      lrow[r] = lrow[r] * a + rs;
      s0[r] = p0;
      s1[r] = p1;
    }

    // ---- rescale running O by alpha (per row) ----
#pragma unroll
    for (int f = 0; f < 4; ++f)
#pragma unroll
      for (int r = 0; r < 8; ++r) o[f][r] *= alpha[r];

    // ---- transpose P: C-layout -> LDS -> A-layout bf16 ----
#pragma unroll
    for (int r = 0; r < 8; ++r) {
      const int mm = r + 8 * h;
      pbuf[mm * 32 + ln] = s0[r];
      pbuf[mm * 32 + 16 + ln] = s1[r];
    }
    asm volatile("s_wait_dscnt 0" ::: "memory");  // same-wave cross-lane hazard
    BFrag pf;
#pragma unroll
    for (int v = 0; v < 8; ++v) {
      const int k0 = ((v >> 2) << 4) + ((v & 3) << 1) + (h << 3);
      const float lo = pbuf[ln * 32 + k0];
      const float hi = pbuf[ln * 32 + k0 + 1];
      pf.u[v] = pack2bf(lo, hi);
    }

    // ---- O += P * V  (V transposed: row = dh col, contiguous tokens) ----
#pragma unroll
    for (int f = 0; f < 4; ++f) o[f] = wmma_bf16(pf.v, vf[f], o[f]);
  }

  // ---- epilogue: O /= l, store bf16 token-major into values[b][s][h*64+n]
  const int b_ = bh >> 4, hd = bh & 15;
#pragma unroll
  for (int f = 0; f < 4; ++f)
#pragma unroll
    for (int r = 0; r < 8; ++r) {
      const int s = qt * 16 + r + 8 * h;
      const float val = o[f][r] / lrow[r];
      Vals[((size_t)b_ * 2048 + s) * 1024 + hd * 64 + f * 16 + ln] = f2bf(val);
    }
}

// ---------------------------------------------------------------------------
// Output projection: out[m][n] = sum_k Vals[m][k] * Wout[n][k] + b_out[n]
// M=8192, K=1024, N=1024.  fp32 output.  Same ping-pong pipeline as gemm_qkv.
// ---------------------------------------------------------------------------
__global__ void __launch_bounds__(128) gemm_out(
    const unsigned short* __restrict__ A, const unsigned short* __restrict__ W,
    const float* __restrict__ bias, float* __restrict__ out) {
  const int lane = threadIdx.x & 31;
  const int wave = threadIdx.x >> 5;
  const int h = lane >> 4, ln = lane & 15;
  const int mt = blockIdx.x;
  const int nbase = blockIdx.y * 256 + wave * 64;

  const unsigned short* arow = A + (size_t)(mt * 16 + ln) * 1024;
  const unsigned short* brow[4];
#pragma unroll
  for (int f = 0; f < 4; ++f)
    brow[f] = W + (size_t)(nbase + f * 16 + ln) * 1024;

  const v8f zero = {0.f, 0.f, 0.f, 0.f, 0.f, 0.f, 0.f, 0.f};
  v8f acc[4] = {zero, zero, zero, zero};

  v16bf a0 = load_frag(arow, h);
  v16bf b0[4];
#pragma unroll
  for (int f = 0; f < 4; ++f) b0[f] = load_frag(brow[f], h);

  for (int kc = 0; kc < 1024; kc += 64) {
    __builtin_prefetch(arow + kc + 128, 0, 1);
    v16bf a1 = load_frag(arow + kc + 32, h);
    v16bf b1[4];
#pragma unroll
    for (int f = 0; f < 4; ++f) b1[f] = load_frag(brow[f] + kc + 32, h);
#pragma unroll
    for (int f = 0; f < 4; ++f) acc[f] = wmma_bf16(a0, b0[f], acc[f]);

    const int kn = (kc + 64 < 1024) ? kc + 64 : 0;
    a0 = load_frag(arow + kn, h);
#pragma unroll
    for (int f = 0; f < 4; ++f) b0[f] = load_frag(brow[f] + kn, h);
#pragma unroll
    for (int f = 0; f < 4; ++f) acc[f] = wmma_bf16(a1, b1[f], acc[f]);
  }

#pragma unroll
  for (int f = 0; f < 4; ++f) {
    const int n = nbase + f * 16 + ln;
    const float bv = bias[n];
#pragma unroll
    for (int r = 0; r < 8; ++r) {
      const int row = mt * 16 + r + 8 * h;
      out[(size_t)row * 1024 + n] = acc[f][r] + bv;
    }
  }
}

// ---------------------------------------------------------------------------
extern "C" void kernel_launch(void* const* d_in, const int* in_sizes, int n_in,
                              void* d_out, int out_size, void* d_ws,
                              size_t ws_size, hipStream_t stream) {
  const float* x    = (const float*)d_in[0];  // [4,2048,1024]
  const float* Wqkv = (const float*)d_in[1];  // [3072,1024]
  const float* bqkv = (const float*)d_in[2];  // [3072]
  const float* Wout = (const float*)d_in[3];  // [1024,1024]
  const float* bout = (const float*)d_in[4];  // [1024]
  float* out = (float*)d_out;

  // Workspace carve-up (bf16 = 2 bytes). Total ~88 MB.
  char* ws = (char*)d_ws;
  unsigned short* Xb  = (unsigned short*)ws; ws += (size_t)8192 * 1024 * 2;
  unsigned short* Wqb = (unsigned short*)ws; ws += (size_t)3072 * 1024 * 2;
  unsigned short* Wob = (unsigned short*)ws; ws += (size_t)1024 * 1024 * 2;
  unsigned short* Qb  = (unsigned short*)ws; ws += (size_t)64 * 2048 * 64 * 2;
  unsigned short* Kb  = (unsigned short*)ws; ws += (size_t)64 * 2048 * 64 * 2;
  unsigned short* Vt  = (unsigned short*)ws; ws += (size_t)64 * 64 * 2048 * 2;
  unsigned short* Vals = (unsigned short*)ws;  // [8192][1024]

  cvt_bf16<<<4096, 256, 0, stream>>>(x, Xb, 8192 * 1024);
  cvt_bf16<<<1536, 256, 0, stream>>>(Wqkv, Wqb, 3072 * 1024);
  cvt_bf16<<<512, 256, 0, stream>>>(Wout, Wob, 1024 * 1024);

  gemm_qkv<<<dim3(512, 12), 128, 0, stream>>>(Xb, Wqb, bqkv, Qb, Kb, Vt);
  attn<<<dim3(32, 64), 128, 0, stream>>>(Qb, Kb, Vt, Vals);
  gemm_out<<<dim3(512, 4), 128, 0, stream>>>(Vals, Wob, bout, out);
}
